// GATNet_64733747085461
// MI455X (gfx1250) — matrix operand order repaired and checked by reading, not compiled, not run on hardware
//
#include <hip/hip_runtime.h>
#include <hip/hip_bf16.h>
#include <stdint.h>

// ---------------- problem constants ----------------
#define NN      50000
#define EEDGES  800000
#define EE_TOT  (EEDGES + NN)     // edges + self loops
#define NF      512
#define HC      64
#define HH      8
#define F1      (HH * HC)         // 512
#define NC      16
#define NEG_SLOPE 0.2f
#define KBLK1   (NF / 32)         // 16 k-blocks of 32 for layer-1 GEMM

typedef __attribute__((ext_vector_type(16))) __bf16        v16bf;
typedef __attribute__((ext_vector_type(8)))  float         v8f;
typedef __attribute__((ext_vector_type(4)))  unsigned int  v4u;
typedef __attribute__((ext_vector_type(4)))  int           v4i;
typedef __attribute__((ext_vector_type(8)))  int           v8i;

#if defined(__has_builtin)
#  if __has_builtin(__builtin_amdgcn_tensor_load_to_lds) && \
      __has_builtin(__builtin_amdgcn_s_wait_tensorcnt)
#    define USE_TDM 1
#  endif
#endif

static inline int cdiv_i(long long a, long long b) { return (int)((a + b - 1) / b); }

// ---------------- utility kernels ----------------
__global__ void fill_f32(float* p, float v, long long n) {
  long long i = (long long)blockIdx.x * blockDim.x + threadIdx.x;
  if (i < n) p[i] = v;
}

// ---------------- fragment packing (f32 -> bf16, WMMA fragment order) -------
// A-pack layout: [Mtile][Kblk][lane(32)][16 bf16], per ISA 7.12.2 16-bit A 16x32:
//   j -> v=j/2, p=j&1 ; k = kb*32 + (v<4?0:16) + (lane>=16?8:0) + (v&3)*2 + p
//   row = mt*16 + (lane&15)
__global__ void pack_a(const float* __restrict__ in, __bf16* __restrict__ out,
                       int M, int K) {
  long long o = (long long)blockIdx.x * blockDim.x + threadIdx.x;
  if (o >= (long long)M * K) return;
  const int j    = (int)(o & 15);
  const int lane = (int)((o >> 4) & 31);
  const long long chunk = o >> 9;              // /512 elements per (mt,kb)
  const int KB = K >> 5;
  const int kb = (int)(chunk % KB);
  const int mt = (int)(chunk / KB);
  const int v = j >> 1, p = j & 1;
  const int k = kb * 32 + ((v < 4) ? 0 : 16) + ((lane >= 16) ? 8 : 0) + ((v & 3) << 1) + p;
  const int row = mt * 16 + (lane & 15);
  out[o] = (__bf16)in[(size_t)row * K + k];
}

// B-pack layout: [Kblk][Ntile][lane(32)][16 bf16], per ISA 16-bit B 32x16:
//   k = kb*32 + (v<4?0:16) + (lane>=16?8:0) + (v&3)*2 + p ; col = nt*16 + (lane&15)
__global__ void pack_b(const float* __restrict__ in, __bf16* __restrict__ out,
                       int K, int Nc) {
  long long o = (long long)blockIdx.x * blockDim.x + threadIdx.x;
  if (o >= (long long)K * Nc) return;
  const int j    = (int)(o & 15);
  const int lane = (int)((o >> 4) & 31);
  const long long chunk = o >> 9;
  const int Nt = Nc >> 4;
  const int nt = (int)(chunk % Nt);
  const int kb = (int)(chunk / Nt);
  const int v = j >> 1, p = j & 1;
  const int k   = kb * 32 + ((v < 4) ? 0 : 16) + ((lane >= 16) ? 8 : 0) + ((v & 3) << 1) + p;
  const int col = nt * 16 + (lane & 15);
  out[o] = (__bf16)in[(size_t)k * Nc + col];
}

// ---------------- TDM / cooperative stage of one 16KB A tile-row -----------
__device__ __forceinline__ void tdm_load_1d_2B(const void* gsrc, unsigned lds_byte,
                                               unsigned nelem /* 2-byte elems */) {
#ifdef USE_TDM
  const unsigned long long ga = (unsigned long long)(uintptr_t)gsrc;
  v4u g0;
  g0[0] = 1u;                                        // count=1, user mode, no gather
  g0[1] = lds_byte;                                  // lds_addr
  g0[2] = (unsigned)(ga & 0xFFFFFFFFu);              // global_addr[31:0]
  g0[3] = (unsigned)((ga >> 32) & 0x01FFFFFFu) | (2u << 30);  // addr[56:32] | type=2
  v8i g1;
  g1[0] = (int)(1u << 16);                           // data_size=1 (2 bytes)
  g1[1] = (int)((nelem & 0xFFFFu) << 16);            // tensor_dim0[15:0]
  g1[2] = (int)(((nelem >> 16) & 0xFFFFu) | (1u << 16)); // tensor_dim0 hi | tensor_dim1=1
  g1[3] = (int)((nelem & 0xFFFFu) << 16);            // tile_dim0 (fits 16b: 8192)
  g1[4] = 1;                                         // tile_dim1=1, tile_dim2=0
  g1[5] = (int)nelem;                                // tensor_dim0_stride lo
  g1[6] = 0;
  g1[7] = 0;
  v4i z4 = {0, 0, 0, 0};
#if __clang_major__ >= 23
  v8i z8 = {0, 0, 0, 0, 0, 0, 0, 0};
  __builtin_amdgcn_tensor_load_to_lds(g0, g1, z4, z4, z8, 0);
#else
  __builtin_amdgcn_tensor_load_to_lds(g0, g1, z4, z4, 0);
#endif
#endif
}

// ---------------- blocked bf16 WMMA GEMM (layer 1: K=512, Nc=512) ----------
// Block = 256 threads (8 waves). Block computes one 16-row M-tile x 256 cols.
// Whole packed A tile-row (KBLK1*32*16 bf16 = 16KB) staged to LDS (TDM or
// cooperative b128 copy); each wave owns 2 N-tiles, 2 WMMAs per k-step.
__global__ __launch_bounds__(256) void gemm1_wmma_blocked(
    const __bf16* __restrict__ Apack, const __bf16* __restrict__ Bpack,
    float* __restrict__ C) {
  __shared__ __align__(16) __bf16 lds_a[KBLK1 * 32 * 16];   // 16 KB
  const int tm   = blockIdx.x;                 // M tile (0..3124)
  const int wave = threadIdx.x >> 5;
  const int lane = threadIdx.x & 31;
  const int Nt   = F1 >> 4;                    // 32 N tiles
  const int tn0  = blockIdx.y * 16 + wave * 2; // this wave's first N tile

  const __bf16* gA = Apack + (size_t)tm * KBLK1 * 512;

#ifdef USE_TDM
  if (wave == 0) {
    tdm_load_1d_2B(gA, (unsigned)(uintptr_t)lds_a, KBLK1 * 512u);
    __builtin_amdgcn_s_wait_tensorcnt(0);
  }
#else
  // cooperative copy: 1024 x 16B chunks
  for (int c = threadIdx.x; c < KBLK1 * 32 * 2; c += blockDim.x) {
    *(uint4*)((char*)lds_a + (size_t)c * 16) =
        *(const uint4*)((const char*)gA + (size_t)c * 16);
  }
#endif
  __syncthreads();

  const v16bf* Bf = (const v16bf*)Bpack;
  v8f acc0 = {}, acc1 = {};
#pragma unroll 4
  for (int kb = 0; kb < KBLK1; ++kb) {
    const v16bf a  = *(const v16bf*)&lds_a[(size_t)(kb * 32 + lane) * 16];
    const v16bf b0 = Bf[((size_t)kb * Nt + tn0) * 32 + lane];
    const v16bf b1 = Bf[((size_t)kb * Nt + tn0 + 1) * 32 + lane];
    acc0 = __builtin_amdgcn_wmma_f32_16x16x32_bf16(false, a, false, b0,
                                                   (short)0, acc0, false, false);
    acc1 = __builtin_amdgcn_wmma_f32_16x16x32_bf16(false, a, false, b1,
                                                   (short)0, acc1, false, false);
  }

  // D layout: lane<16 -> rows r, cols lane; lane>=16 -> rows 8+r, cols lane-16
  const int mbase = tm * 16 + ((lane >> 4) ? 8 : 0);
  const int n0    = tn0 * 16 + (lane & 15);
#pragma unroll
  for (int r = 0; r < 8; ++r) {
    C[(size_t)(mbase + r) * F1 + n0]      = acc0[r];
    C[(size_t)(mbase + r) * F1 + n0 + 16] = acc1[r];
  }
}

// ---------------- simple wave-per-tile GEMM (layer 2: Nc=16) ---------------
__global__ void gemm_wmma_simple(const __bf16* __restrict__ Apack,
                                 const __bf16* __restrict__ Bpack,
                                 float* __restrict__ C,
                                 int M, int K, int Nc) {
  const int wave = (int)((blockIdx.x * (long long)blockDim.x + threadIdx.x) >> 5);
  const int lane = threadIdx.x & 31;
  const int KB = K >> 5, Nt = Nc >> 4;
  const int tm = wave / Nt, tn = wave % Nt;
  if (tm * 16 >= M) return;                       // wave-uniform exit

  const v16bf* Af = (const v16bf*)Apack;
  const v16bf* Bf = (const v16bf*)Bpack;
  v8f acc = {};
#pragma unroll 4
  for (int kb = 0; kb < KB; ++kb) {
    const v16bf a = Af[((size_t)tm * KB + kb) * 32 + lane];
    const v16bf b = Bf[((size_t)kb * Nt + tn) * 32 + lane];
    acc = __builtin_amdgcn_wmma_f32_16x16x32_bf16(false, a, false, b,
                                                  (short)0, acc, false, false);
  }
  const int mbase = tm * 16 + ((lane >> 4) ? 8 : 0);
  const int n     = tn * 16 + (lane & 15);
#pragma unroll
  for (int r = 0; r < 8; ++r) C[(size_t)(mbase + r) * Nc + n] = acc[r];
}

// ---------------- per-node attention coefficients ----------------
__global__ void alpha_kernel(const float* __restrict__ hmat,
                             const float* __restrict__ a_src,
                             const float* __restrict__ a_dst,
                             float* __restrict__ asrc, float* __restrict__ adst,
                             int n_nodes, int heads, int ch) {
  long long i = (long long)blockIdx.x * blockDim.x + threadIdx.x;
  if (i >= (long long)n_nodes * heads) return;
  const int node = (int)(i / heads);
  const int hd   = (int)(i % heads);
  const float* hp = hmat + (size_t)node * heads * ch + (size_t)hd * ch;
  const float* as = a_src + (size_t)hd * ch;
  const float* ad = a_dst + (size_t)hd * ch;
  float s = 0.f, d = 0.f;
  for (int c = 0; c < ch; ++c) { float v = hp[c]; s += v * as[c]; d += v * ad[c]; }
  asrc[i] = s;
  adst[i] = d;
}

// ---------------- edge kernels ----------------
__device__ __forceinline__ void edge_sd(const long long* __restrict__ ei,
                                        long long e, int& s, int& d) {
  if (e < (long long)EEDGES) { s = (int)ei[e]; d = (int)ei[(long long)EEDGES + e]; }
  else { s = d = (int)(e - (long long)EEDGES); }
}

__device__ __forceinline__ float leaky(float v) { return v > 0.f ? v : NEG_SLOPE * v; }

__device__ __forceinline__ float atomicMaxF(float* addr, float v) {
  unsigned int* ua = (unsigned int*)addr;
  unsigned int old = *ua, assumed;
  do {
    assumed = old;
    if (__uint_as_float(assumed) >= v) break;
    old = atomicCAS(ua, assumed, __float_as_uint(v));
  } while (old != assumed);
  return __uint_as_float(old);
}

__global__ void edge_max(const long long* __restrict__ ei,
                         const float* __restrict__ asrc, const float* __restrict__ adst,
                         float* __restrict__ m, int heads) {
  long long i = (long long)blockIdx.x * blockDim.x + threadIdx.x;
  if (i >= (long long)EE_TOT * heads) return;
  const long long e = i / heads;
  const int hd = (int)(i % heads);
  int s, d; edge_sd(ei, e, s, d);
  const float ev = leaky(asrc[(size_t)s * heads + hd] + adst[(size_t)d * heads + hd]);
  atomicMaxF(&m[(size_t)d * heads + hd], ev);
}

__global__ void edge_expsum(const long long* __restrict__ ei,
                            const float* __restrict__ asrc, const float* __restrict__ adst,
                            const float* __restrict__ m,
                            float* __restrict__ ex_buf, float* __restrict__ denom,
                            int heads) {
  long long i = (long long)blockIdx.x * blockDim.x + threadIdx.x;
  if (i >= (long long)EE_TOT * heads) return;
  const long long e = i / heads;
  const int hd = (int)(i % heads);
  int s, d; edge_sd(ei, e, s, d);
  const float ev = leaky(asrc[(size_t)s * heads + hd] + adst[(size_t)d * heads + hd]);
  const float ex = __expf(ev - m[(size_t)d * heads + hd]);
  ex_buf[i] = ex;
  atomicAdd(&denom[(size_t)d * heads + hd], ex);
}

__global__ void edge_aggregate(const long long* __restrict__ ei,
                               const float* __restrict__ ex_buf,
                               const float* __restrict__ denom,
                               const float* __restrict__ hmat,
                               float* __restrict__ agg,
                               int heads, int ch) {
  long long i = (long long)blockIdx.x * blockDim.x + threadIdx.x;
  if (i >= (long long)EE_TOT * heads * ch) return;
  const int c = (int)(i % ch);
  long long t = i / ch;
  const int hd = (int)(t % heads);
  const long long e = t / heads;
  int s, d; edge_sd(ei, e, s, d);
  const float alpha = ex_buf[e * heads + hd] / denom[(size_t)d * heads + hd];
  const float val = hmat[(size_t)s * heads * ch + (size_t)hd * ch + c] * alpha;
  atomicAdd(&agg[(size_t)d * heads * ch + (size_t)hd * ch + c], val);
}

// emb = agg + b -> d_out (f32); hact_f32 = elu(emb) (packed to bf16 afterwards)
__global__ void bias_elu(const float* __restrict__ agg, const float* __restrict__ b,
                         float* __restrict__ emb_out, float* __restrict__ hact,
                         long long n, int F) {
  long long i = (long long)blockIdx.x * blockDim.x + threadIdx.x;
  if (i >= n) return;
  const float v = agg[i] + b[i % F];
  emb_out[i] = v;
  hact[i] = v > 0.f ? v : (__expf(v) - 1.f);
}

__global__ void bias_add(const float* __restrict__ agg, const float* __restrict__ b,
                         float* __restrict__ out, long long n, int F) {
  long long i = (long long)blockIdx.x * blockDim.x + threadIdx.x;
  if (i >= n) return;
  out[i] = agg[i] + b[i % F];
}

// ---------------- launcher ----------------
extern "C" void kernel_launch(void* const* d_in, const int* in_sizes, int n_in,
                              void* d_out, int out_size, void* d_ws, size_t ws_size,
                              hipStream_t stream) {
  (void)in_sizes; (void)n_in; (void)out_size; (void)ws_size;

  const float*     x      = (const float*)d_in[0];
  const long long* ei     = (const long long*)d_in[1];   // int64 [2, E]
  const float*     w1     = (const float*)d_in[2];       // [512, 512]
  const float*     a_src1 = (const float*)d_in[3];
  const float*     a_dst1 = (const float*)d_in[4];
  const float*     b1     = (const float*)d_in[5];
  const float*     w2     = (const float*)d_in[6];       // [512, 16]
  const float*     a_src2 = (const float*)d_in[7];
  const float*     a_dst2 = (const float*)d_in[8];
  const float*     b2     = (const float*)d_in[9];

  float* logits_out = (float*)d_out;                     // [N, 16]
  float* emb_out    = logits_out + (size_t)NN * NC;      // [N, 512]

  size_t off = 0;
  auto carve = [&](size_t bytes) -> void* {
    void* p = (char*)d_ws + off;
    off += (bytes + 255) & ~(size_t)255;
    return p;
  };
  __bf16* xpack    = (__bf16*)carve((size_t)NN * NF * 2);
  __bf16* w1pack   = (__bf16*)carve((size_t)NF * F1 * 2);
  float*  h1       = (float*) carve((size_t)NN * F1 * 4);
  float*  asrc1    = (float*) carve((size_t)NN * HH * 4);
  float*  adst1    = (float*) carve((size_t)NN * HH * 4);
  float*  m1       = (float*) carve((size_t)NN * HH * 4);
  float*  denom1   = (float*) carve((size_t)NN * HH * 4);
  float*  ex1      = (float*) carve((size_t)EE_TOT * HH * 4);
  float*  agg1     = (float*) carve((size_t)NN * F1 * 4);
  float*  hact_f32 = (float*) carve((size_t)NN * F1 * 4);
  __bf16* hactpack = (__bf16*)carve((size_t)NN * F1 * 2);
  __bf16* w2pack   = (__bf16*)carve((size_t)F1 * NC * 2);
  float*  h2       = (float*) carve((size_t)NN * NC * 4);
  float*  asrc2    = (float*) carve((size_t)NN * 4);
  float*  adst2    = (float*) carve((size_t)NN * 4);
  float*  m2       = (float*) carve((size_t)NN * 4);
  float*  denom2   = (float*) carve((size_t)NN * 4);
  float*  ex2      = (float*) carve((size_t)EE_TOT * 4);
  float*  agg2     = (float*) carve((size_t)NN * NC * 4);

  const int T = 256;
  const float NEG_INF = -3.4e38f;

  // ---- layer 1 ----
  pack_a<<<cdiv_i((long long)NN * NF, T), T, 0, stream>>>(x, xpack, NN, NF);
  pack_b<<<cdiv_i((long long)NF * F1, T), T, 0, stream>>>(w1, w1pack, NF, F1);

  gemm1_wmma_blocked<<<dim3(NN / 16, F1 / 256), T, 0, stream>>>(xpack, w1pack, h1);

  alpha_kernel<<<cdiv_i((long long)NN * HH, T), T, 0, stream>>>(h1, a_src1, a_dst1,
                                                                asrc1, adst1, NN, HH, HC);

  fill_f32<<<cdiv_i((long long)NN * HH, T), T, 0, stream>>>(m1, NEG_INF, (long long)NN * HH);
  fill_f32<<<cdiv_i((long long)NN * HH, T), T, 0, stream>>>(denom1, 0.f, (long long)NN * HH);
  fill_f32<<<cdiv_i((long long)NN * F1, T), T, 0, stream>>>(agg1, 0.f, (long long)NN * F1);

  edge_max<<<cdiv_i((long long)EE_TOT * HH, T), T, 0, stream>>>(ei, asrc1, adst1, m1, HH);
  edge_expsum<<<cdiv_i((long long)EE_TOT * HH, T), T, 0, stream>>>(ei, asrc1, adst1, m1,
                                                                   ex1, denom1, HH);
  edge_aggregate<<<cdiv_i((long long)EE_TOT * HH * HC, T), T, 0, stream>>>(ei, ex1, denom1,
                                                                            h1, agg1, HH, HC);

  bias_elu<<<cdiv_i((long long)NN * F1, T), T, 0, stream>>>(agg1, b1, emb_out, hact_f32,
                                                            (long long)NN * F1, F1);

  // ---- layer 2 ----
  pack_a<<<cdiv_i((long long)NN * F1, T), T, 0, stream>>>(hact_f32, hactpack, NN, F1);
  pack_b<<<cdiv_i((long long)F1 * NC, T), T, 0, stream>>>(w2, w2pack, F1, NC);

  {
    const long long waves = (long long)(NN / 16) * (NC / 16);   // 3125
    gemm_wmma_simple<<<cdiv_i(waves * 32, T), T, 0, stream>>>(hactpack, w2pack, h2,
                                                              NN, F1, NC);
  }

  alpha_kernel<<<cdiv_i((long long)NN, T), T, 0, stream>>>(h2, a_src2, a_dst2,
                                                           asrc2, adst2, NN, 1, NC);

  fill_f32<<<cdiv_i((long long)NN, T), T, 0, stream>>>(m2, NEG_INF, (long long)NN);
  fill_f32<<<cdiv_i((long long)NN, T), T, 0, stream>>>(denom2, 0.f, (long long)NN);
  fill_f32<<<cdiv_i((long long)NN * NC, T), T, 0, stream>>>(agg2, 0.f, (long long)NN * NC);

  edge_max<<<cdiv_i((long long)EE_TOT, T), T, 0, stream>>>(ei, asrc2, adst2, m2, 1);
  edge_expsum<<<cdiv_i((long long)EE_TOT, T), T, 0, stream>>>(ei, asrc2, adst2, m2,
                                                              ex2, denom2, 1);
  edge_aggregate<<<cdiv_i((long long)EE_TOT * NC, T), T, 0, stream>>>(ei, ex2, denom2,
                                                                       h2, agg2, 1, NC);

  bias_add<<<cdiv_i((long long)NN * NC, T), T, 0, stream>>>(agg2, b2, logits_out,
                                                            (long long)NN * NC, NC);
}
